// QuantBertGatedLinearUnitMLP_37726992728252
// MI455X (gfx1250) — 8-bit matrix operand order repaired and checked by reading, compiled, not run on hardware
//
#include <hip/hip_runtime.h>

#define TOK   8192
#define HDIM  1024
#define IDIM  4096
#define KC    128

typedef __attribute__((ext_vector_type(8))) int v8i;

__device__ __forceinline__ void async_copy16(unsigned lds_off, const signed char* gaddr) {
  asm volatile("global_load_async_to_lds_b128 %0, %1, off"
               :: "v"(lds_off), "v"(gaddr) : "memory");
}
__device__ __forceinline__ void wait_async0() {
  asm volatile("s_wait_asynccnt 0" ::: "memory");
}

__global__ __launch_bounds__(256) void quant_rows_kernel(
    const float* __restrict__ in, signed char* __restrict__ q,
    float* __restrict__ scale, int K) {
  __shared__ float red[256];
  const int row = blockIdx.x;
  const float4* x4 = (const float4*)(in + (size_t)row * K);
  const int K4 = K >> 2;
  float amax = 0.0f;
  for (int k = threadIdx.x; k < K4; k += 256) {
    const float4 v = x4[k];
    amax = fmaxf(amax, fmaxf(fmaxf(fabsf(v.x), fabsf(v.y)), fmaxf(fabsf(v.z), fabsf(v.w))));
  }
  red[threadIdx.x] = amax;
  __syncthreads();
  for (int s = 128; s > 0; s >>= 1) {
    if (threadIdx.x < (unsigned)s) red[threadIdx.x] = fmaxf(red[threadIdx.x], red[threadIdx.x + s]);
    __syncthreads();
  }
  const float sc = fmaxf(red[0] * (1.0f / 127.0f), 1e-8f);
  if (threadIdx.x == 0) scale[row] = sc;
  const float inv = 1.0f / sc;
  char4* q4 = (char4*)(q + (size_t)row * K);
  for (int k = threadIdx.x; k < K4; k += 256) {
    const float4 v = x4[k];
    char4 c;
    c.x = (signed char)fminf(fmaxf(rintf(v.x * inv), -128.0f), 127.0f);
    c.y = (signed char)fminf(fmaxf(rintf(v.y * inv), -128.0f), 127.0f);
    c.z = (signed char)fminf(fmaxf(rintf(v.z * inv), -128.0f), 127.0f);
    c.w = (signed char)fminf(fmaxf(rintf(v.w * inv), -128.0f), 127.0f);
    q4[k] = c;
  }
}

__device__ __forceinline__ v8i frag_a(const unsigned int* __restrict__ tile,
                                      int rowBase, int lane, int h) {
  const int r = (rowBase + (lane & 15)) * 32 + h * 16 + ((lane >> 4) << 1);
  v8i a;
  a[0] = (int)tile[r + 0];  a[1] = (int)tile[r + 1];
  a[2] = (int)tile[r + 4];  a[3] = (int)tile[r + 5];
  a[4] = (int)tile[r + 8];  a[5] = (int)tile[r + 9];
  a[6] = (int)tile[r + 12]; a[7] = (int)tile[r + 13];
  return a;
}

__device__ __forceinline__ v8i frag_b(const unsigned int* __restrict__ tile,
                                      int rowBase, int lane, int h) {
  const int r = (rowBase + (lane & 15)) * 32 + h * 16 + ((lane >> 4) << 1);
  v8i b;
  b[0] = (int)tile[r + 0];  b[1] = (int)tile[r + 1];
  b[2] = (int)tile[r + 4];  b[3] = (int)tile[r + 5];
  b[4] = (int)tile[r + 8];  b[5] = (int)tile[r + 9];
  b[6] = (int)tile[r + 12]; b[7] = (int)tile[r + 13];
  return b;
}

__global__ __launch_bounds__(256) void gemm1_glu_kernel(
    const signed char* __restrict__ xq, const float* __restrict__ sx,
    const signed char* __restrict__ wq, const float* __restrict__ sw,
    const float* __restrict__ bias, float* __restrict__ h2) {
  __shared__ unsigned int As[2][64 * 32];
  __shared__ unsigned int Bg[2][64 * 32];
  __shared__ unsigned int Bn[2][64 * 32];

  const int tid  = threadIdx.x;
  const int wave = tid >> 5;
  const int lane = tid & 31;
  const int wm   = wave >> 1;
  const int wn   = wave & 1;
  const int m0   = blockIdx.y * 64;
  const int n0   = blockIdx.x * 64;

  const int ldRow = tid >> 2;
  const int ldSeg = (tid & 3) * 32;

  v8i accg0 = {}, accn0 = {}, accg1 = {}, accn1 = {};

  const signed char* gA = xq + (size_t)(m0 + ldRow) * HDIM + ldSeg;
  const signed char* gG = wq + (size_t)(n0 + ldRow) * HDIM + ldSeg;
  const signed char* gN = wq + (size_t)(n0 + ldRow + IDIM) * HDIM + ldSeg;

  auto issue_stage = [&](int kb, int buf) {
    const unsigned la = (unsigned)(size_t)&As[buf][ldRow * 32] + ldSeg;
    const unsigned lg = (unsigned)(size_t)&Bg[buf][ldRow * 32] + ldSeg;
    const unsigned ln = (unsigned)(size_t)&Bn[buf][ldRow * 32] + ldSeg;
    async_copy16(la,      gA + kb);
    async_copy16(la + 16, gA + kb + 16);
    async_copy16(lg,      gG + kb);
    async_copy16(lg + 16, gG + kb + 16);
    async_copy16(ln,      gN + kb);
    async_copy16(ln + 16, gN + kb + 16);
  };

  issue_stage(0, 0);
  const int STAGES = HDIM / KC;
  for (int s = 0; s < STAGES; ++s) {
    const int cur = s & 1;
    wait_async0();
    __syncthreads();
    if (s + 1 < STAGES) issue_stage((s + 1) * KC, cur ^ 1);

#pragma unroll
    for (int h = 0; h < 2; ++h) {
      const v8i a   = frag_a(&As[cur][0], wm * 16, lane, h);
      const v8i b0g = frag_b(&Bg[cur][0], wn * 32 + 0,  lane, h);
      const v8i b0n = frag_b(&Bn[cur][0], wn * 32 + 0,  lane, h);
      const v8i b1g = frag_b(&Bg[cur][0], wn * 32 + 16, lane, h);
      const v8i b1n = frag_b(&Bn[cur][0], wn * 32 + 16, lane, h);
      accg0 = __builtin_amdgcn_wmma_i32_16x16x64_iu8(true, a, true, b0g, accg0, false, false);
      accn0 = __builtin_amdgcn_wmma_i32_16x16x64_iu8(true, a, true, b0n, accn0, false, false);
      accg1 = __builtin_amdgcn_wmma_i32_16x16x64_iu8(true, a, true, b1g, accg1, false, false);
      accn1 = __builtin_amdgcn_wmma_i32_16x16x64_iu8(true, a, true, b1n, accn1, false, false);
    }
  }

  const int fn = lane & 15;
  const int mh = (lane >> 4) * 8;
#pragma unroll
  for (int r = 0; r < 8; ++r) {
    const int m = m0 + wm * 16 + mh + r;
    const float sxm = sx[m];
    {
      const int j = n0 + wn * 32 + fn;
      const float g = (float)accg0[r] * (sxm * sw[j]) + bias[j];
      const float n = (float)accn0[r] * (sxm * sw[j + IDIM]) + bias[j + IDIM];
      h2[(size_t)m * IDIM + j] = (g / (1.0f + __expf(-g))) * n;
    }
    {
      const int j = n0 + wn * 32 + 16 + fn;
      const float g = (float)accg1[r] * (sxm * sw[j]) + bias[j];
      const float n = (float)accn1[r] * (sxm * sw[j + IDIM]) + bias[j + IDIM];
      h2[(size_t)m * IDIM + j] = (g / (1.0f + __expf(-g))) * n;
    }
  }
}

__global__ __launch_bounds__(256) void gemm2_kernel(
    const signed char* __restrict__ aq, const float* __restrict__ sa,
    const signed char* __restrict__ wq, const float* __restrict__ sw,
    const float* __restrict__ bias, const float* __restrict__ residual,
    float* __restrict__ xout) {
  __shared__ unsigned int As[2][64 * 32];
  __shared__ unsigned int Bs[2][64 * 32];

  const int tid  = threadIdx.x;
  const int wave = tid >> 5;
  const int lane = tid & 31;
  const int wm   = wave >> 1;
  const int wn   = wave & 1;
  const int m0   = blockIdx.y * 64;
  const int n0   = blockIdx.x * 64;

  const int ldRow = tid >> 2;
  const int ldSeg = (tid & 3) * 32;

  v8i acc0 = {}, acc1 = {};

  const signed char* gA = aq + (size_t)(m0 + ldRow) * IDIM + ldSeg;
  const signed char* gB = wq + (size_t)(n0 + ldRow) * IDIM + ldSeg;

  auto issue_stage = [&](int kb, int buf) {
    const unsigned la = (unsigned)(size_t)&As[buf][ldRow * 32] + ldSeg;
    const unsigned lb = (unsigned)(size_t)&Bs[buf][ldRow * 32] + ldSeg;
    async_copy16(la,      gA + kb);
    async_copy16(la + 16, gA + kb + 16);
    async_copy16(lb,      gB + kb);
    async_copy16(lb + 16, gB + kb + 16);
  };

  issue_stage(0, 0);
  const int STAGES = IDIM / KC;
  for (int s = 0; s < STAGES; ++s) {
    const int cur = s & 1;
    wait_async0();
    __syncthreads();
    if (s + 1 < STAGES) issue_stage((s + 1) * KC, cur ^ 1);

#pragma unroll
    for (int h = 0; h < 2; ++h) {
      const v8i a  = frag_a(&As[cur][0], wm * 16, lane, h);
      const v8i b0 = frag_b(&Bs[cur][0], wn * 32 + 0,  lane, h);
      const v8i b1 = frag_b(&Bs[cur][0], wn * 32 + 16, lane, h);
      acc0 = __builtin_amdgcn_wmma_i32_16x16x64_iu8(true, a, true, b0, acc0, false, false);
      acc1 = __builtin_amdgcn_wmma_i32_16x16x64_iu8(true, a, true, b1, acc1, false, false);
    }
  }

  const int fn = lane & 15;
  const int mh = (lane >> 4) * 8;
#pragma unroll
  for (int r = 0; r < 8; ++r) {
    const int m = m0 + wm * 16 + mh + r;
    const float sam = sa[m];
    {
      const int j = n0 + wn * 32 + fn;
      xout[(size_t)m * HDIM + j] =
          (float)acc0[r] * (sam * sw[j]) + bias[j] + residual[(size_t)m * HDIM + j];
    }
    {
      const int j = n0 + wn * 32 + 16 + fn;
      xout[(size_t)m * HDIM + j] =
          (float)acc1[r] * (sam * sw[j]) + bias[j] + residual[(size_t)m * HDIM + j];
    }
  }
}

__global__ __launch_bounds__(256) void ln_rows_kernel(
    const float* __restrict__ x, const float* __restrict__ gamma,
    const float* __restrict__ beta, float* __restrict__ out) {
  __shared__ float rs[256];
  __shared__ float rq[256];
  const int row = blockIdx.x;
  const float4* x4 = (const float4*)(x + (size_t)row * HDIM);
  float s = 0.0f, s2 = 0.0f;
  for (int k = threadIdx.x; k < HDIM / 4; k += 256) {
    const float4 v = x4[k];
    s  += v.x + v.y + v.z + v.w;
    s2 += v.x * v.x + v.y * v.y + v.z * v.z + v.w * v.w;
  }
  rs[threadIdx.x] = s; rq[threadIdx.x] = s2;
  __syncthreads();
  for (int t = 128; t > 0; t >>= 1) {
    if (threadIdx.x < (unsigned)t) {
      rs[threadIdx.x] += rs[threadIdx.x + t];
      rq[threadIdx.x] += rq[threadIdx.x + t];
    }
    __syncthreads();
  }
  const float mu  = rs[0] * (1.0f / HDIM);
  const float var = rq[0] * (1.0f / HDIM) - mu * mu;
  const float inv = rsqrtf(var + 1e-12f);
  const float4* g4 = (const float4*)gamma;
  const float4* b4 = (const float4*)beta;
  float4* o4 = (float4*)(out + (size_t)row * HDIM);
  for (int k = threadIdx.x; k < HDIM / 4; k += 256) {
    const float4 v = x4[k], gg = g4[k], bb = b4[k];
    float4 o;
    o.x = (v.x - mu) * inv * gg.x + bb.x;
    o.y = (v.y - mu) * inv * gg.y + bb.y;
    o.z = (v.z - mu) * inv * gg.z + bb.z;
    o.w = (v.w - mu) * inv * gg.w + bb.w;
    o4[k] = o;
  }
}

extern "C" void kernel_launch(void* const* d_in, const int* in_sizes, int n_in,
                              void* d_out, int out_size, void* d_ws, size_t ws_size,
                              hipStream_t stream) {
  const float* hidden = (const float*)d_in[0];
  const float* Wg     = (const float*)d_in[1];
  const float* bg     = (const float*)d_in[2];
  const float* Wo     = (const float*)d_in[3];
  const float* bo     = (const float*)d_in[4];
  const float* gamma  = (const float*)d_in[5];
  const float* beta   = (const float*)d_in[6];
  float* out = (float*)d_out;

  char* ws = (char*)d_ws;
  size_t off = 0;
  auto carve = [&](size_t bytes) -> void* {
    off = (off + 255) & ~(size_t)255;
    void* p = ws + off;
    off += bytes;
    return p;
  };

  signed char* xq  = (signed char*)carve((size_t)TOK * HDIM);
  float*       sx  = (float*)carve((size_t)TOK * sizeof(float));
  signed char* wgq = (signed char*)carve((size_t)2 * IDIM * HDIM);
  float*       swg = (float*)carve((size_t)2 * IDIM * sizeof(float));
  signed char* woq = (signed char*)carve((size_t)HDIM * IDIM);
  float*       swo = (float*)carve((size_t)HDIM * sizeof(float));
  float*       h2  = (float*)carve((size_t)TOK * IDIM * sizeof(float));
  signed char* h2q = (signed char*)carve((size_t)TOK * IDIM);
  float*       sh2 = (float*)carve((size_t)TOK * sizeof(float));
  float*       xln = (float*)carve((size_t)TOK * HDIM * sizeof(float));

  quant_rows_kernel<<<TOK, 256, 0, stream>>>(hidden, xq, sx, HDIM);
  quant_rows_kernel<<<2 * IDIM, 256, 0, stream>>>(Wg, wgq, swg, HDIM);
  quant_rows_kernel<<<HDIM, 256, 0, stream>>>(Wo, woq, swo, IDIM);

  {
    dim3 grid(IDIM / 64, TOK / 64);
    gemm1_glu_kernel<<<grid, 256, 0, stream>>>(xq, sx, wgq, swg, bg, h2);
  }

  quant_rows_kernel<<<TOK, 256, 0, stream>>>(h2, h2q, sh2, IDIM);

  {
    dim3 grid(HDIM / 64, TOK / 64);
    gemm2_kernel<<<grid, 256, 0, stream>>>(h2q, sh2, woq, swo, bo, hidden, xln);
  }

  ln_rows_kernel<<<TOK, 256, 0, stream>>>(xln, gamma, beta, out);
}
